// GCNnet_volleyball_6476810682381
// MI455X (gfx1250) — matrix operand order repaired and checked by reading, not compiled, not run on hardware
//
#include <hip/hip_runtime.h>
#include <hip/hip_bf16.h>
#include <math.h>

typedef __attribute__((ext_vector_type(16))) __bf16 v16bf;
typedef __attribute__((ext_vector_type(8)))  __bf16 v8bf;
typedef __attribute__((ext_vector_type(8)))  float  v8f;

#define BT   4096
#define NTOK 12
#define NFG_ 1024
#define NFR_ 256
#define XS   1032   /* bf16 LDS row stride (2064B -> 4-bank skew per row) */
#define TS   520    /* f32 LDS row stride for theta|phi */
#define THR  31.4f  /* 0.2 * 157 */

union ABf { v16bf v; v8bf h[2]; };

// A-fragment (16x32 bf16, M x K), contiguous K order matching B:
// lanes 0-15: K=0..15(+kk), lanes16-31: K=16..31(+kk)
// -> two contiguous 16B loads at +ahi and +8+ahi.
__device__ __forceinline__ v16bf load_a_frag(const __bf16* p) {
  ABf a;
  a.h[0] = *(const v8bf*)p;
  a.h[1] = *(const v8bf*)(p + 8);
  return a.v;
}

__global__ void wconv_kernel(const float* __restrict__ Wt, const float* __restrict__ Wp,
                             const float* __restrict__ Wg,
                             __bf16* __restrict__ wcat, __bf16* __restrict__ wgb) {
  int idx = blockIdx.x * blockDim.x + threadIdx.x;
  if (idx < 512 * 1024) {
    int r = idx >> 10, c = idx & 1023;
    float v = (r < 256) ? Wt[r * 1024 + c] : Wp[(r - 256) * 1024 + c];
    wcat[idx] = (__bf16)v;
  }
  if (idx < 1024 * 1024) wgb[idx] = (__bf16)Wg[idx];
}

__launch_bounds__(512)
__global__ void gcn_fused_kernel(const float* __restrict__ x, const float* __restrict__ boxes,
                                 const float* __restrict__ bth, const float* __restrict__ bph,
                                 const __bf16* __restrict__ wcat, const __bf16* __restrict__ wgb,
                                 float* __restrict__ out, float* __restrict__ rgout) {
  __shared__ __align__(32) __bf16 xb[16 * XS];
  __shared__ __align__(32) __bf16 aggb[16 * XS];
  __shared__ __align__(16) float  tp[16 * TS];
  __shared__ float sim[NTOK * NTOK];
  __shared__ float rg[NTOK * NTOK];
  __shared__ float cx[NTOK], cy[NTOK];

  const int b    = blockIdx.x;
  const int tid  = threadIdx.x;
  const int wv   = tid >> 5;
  const int lane = tid & 31;
  const int lrow = lane & 15;
  const int hi   = (lane & 16) ? 1 : 0;
  const int ahi  = hi * 16;   // A-frag K shift for high half-wave
  const int bhi  = hi * 16;   // B-frag K shift for high half-wave
  const int mbase = hi * 8;   // C/D: VGPR i -> M = mbase + i

  // ---- phase 0: stage x[b] -> bf16 LDS, zero pad rows, box centers ----
  const float* xg = x + (size_t)b * NTOK * NFG_;
  for (int idx = tid; idx < NTOK * NFG_; idx += 512) {
    int r = idx >> 10, c = idx & 1023;
    xb[r * XS + c] = (__bf16)xg[idx];
  }
  for (int idx = tid; idx < 4 * NFG_; idx += 512) {
    int r = 12 + (idx >> 10), c = idx & 1023;
    xb[r * XS + c]   = (__bf16)0.0f;
    aggb[r * XS + c] = (__bf16)0.0f;
  }
  if (tid < NTOK) {
    const float* bx = boxes + (size_t)(b * NTOK + tid) * 4;
    cx[tid] = (bx[0] + bx[2]) * 0.5f;
    cy[tid] = (bx[1] + bx[3]) * 0.5f;
  }
  __syncthreads();

  // ---- phase 1: theta|phi = xb @ [W_theta;W_phi]^T  (16 x 512) ----
  {
    v8f acc0 = {0.f,0.f,0.f,0.f,0.f,0.f,0.f,0.f};
    v8f acc1 = acc0;
    const int nt0 = wv * 2, nt1 = wv * 2 + 1;
    const __bf16* w0 = wcat + (size_t)(nt0 * 16 + lrow) * NFG_;
    const __bf16* w1 = wcat + (size_t)(nt1 * 16 + lrow) * NFG_;
    for (int kk = 0; kk < NFG_; kk += 32) {
      v16bf a  = load_a_frag(&xb[lrow * XS + kk + ahi]);
      v16bf b0 = *(const v16bf*)(w0 + kk + bhi);
      v16bf b1 = *(const v16bf*)(w1 + kk + bhi);
      acc0 = __builtin_amdgcn_wmma_f32_16x16x32_bf16(false, a, false, b0, (short)0, acc0, false, false);
      acc1 = __builtin_amdgcn_wmma_f32_16x16x32_bf16(false, a, false, b1, (short)0, acc1, false, false);
    }
    int nc0 = nt0 * 16 + lrow, nc1 = nt1 * 16 + lrow;
    float bias0 = (nc0 < NFR_) ? bth[nc0] : bph[nc0 - NFR_];
    float bias1 = (nc1 < NFR_) ? bth[nc1] : bph[nc1 - NFR_];
#pragma unroll
    for (int i = 0; i < 8; ++i) {
      tp[(mbase + i) * TS + nc0] = acc0[i] + bias0;
      tp[(mbase + i) * TS + nc1] = acc1[i] + bias1;
    }
  }
  __syncthreads();

  // ---- phase 2a: sim = theta.phi^T / 16, distance mask ----
  if (tid < NTOK * NTOK) {
    int i = tid / NTOK, j = tid % NTOK;
    float s = 0.f;
    for (int r = 0; r < NFR_; ++r)
      s += tp[i * TS + r] * tp[j * TS + NFR_ + r];
    s *= (1.0f / 16.0f);
    float dx = cx[i] - cx[j], dy = cy[i] - cy[j];
    float dist = sqrtf(dx * dx + dy * dy);
    sim[tid] = (dist > THR) ? -__builtin_inff() : s;
  }
  __syncthreads();

  // ---- phase 2b: row softmax -> relation graph ----
  if (tid < NTOK) {
    float mx = -__builtin_inff();
    for (int j = 0; j < NTOK; ++j) mx = fmaxf(mx, sim[tid * NTOK + j]);
    float e[NTOK];
    float sum = 0.f;
    for (int j = 0; j < NTOK; ++j) {
      float ev = expf(sim[tid * NTOK + j] - mx);
      e[j] = ev; sum += ev;
    }
    float inv = 1.0f / sum;
    for (int j = 0; j < NTOK; ++j) {
      float v = e[j] * inv;
      rg[tid * NTOK + j] = v;
      rgout[(size_t)b * (NTOK * NTOK) + tid * NTOK + j] = v;
    }
  }
  __syncthreads();

  // ---- phase 3: agg = rg @ x  -> bf16 LDS ----
  for (int idx = tid; idx < NTOK * NFG_; idx += 512) {
    int i = idx >> 10, f = idx & 1023;
    float s = 0.f;
#pragma unroll
    for (int j = 0; j < NTOK; ++j)
      s += rg[i * NTOK + j] * (float)xb[j * XS + f];
    aggb[i * XS + f] = (__bf16)s;
  }
  __syncthreads();

  // ---- phase 4: out = relu(agg @ W_gcn^T)  (16 x 1024) ----
  {
    v8f acc0 = {0.f,0.f,0.f,0.f,0.f,0.f,0.f,0.f};
    v8f acc1 = acc0, acc2 = acc0, acc3 = acc0;
    const int nt = wv * 4;
    const __bf16* w0 = wgb + (size_t)((nt + 0) * 16 + lrow) * NFG_;
    const __bf16* w1 = wgb + (size_t)((nt + 1) * 16 + lrow) * NFG_;
    const __bf16* w2 = wgb + (size_t)((nt + 2) * 16 + lrow) * NFG_;
    const __bf16* w3 = wgb + (size_t)((nt + 3) * 16 + lrow) * NFG_;
    for (int kk = 0; kk < NFG_; kk += 32) {
      v16bf a  = load_a_frag(&aggb[lrow * XS + kk + ahi]);
      v16bf b0 = *(const v16bf*)(w0 + kk + bhi);
      v16bf b1 = *(const v16bf*)(w1 + kk + bhi);
      v16bf b2 = *(const v16bf*)(w2 + kk + bhi);
      v16bf b3 = *(const v16bf*)(w3 + kk + bhi);
      acc0 = __builtin_amdgcn_wmma_f32_16x16x32_bf16(false, a, false, b0, (short)0, acc0, false, false);
      acc1 = __builtin_amdgcn_wmma_f32_16x16x32_bf16(false, a, false, b1, (short)0, acc1, false, false);
      acc2 = __builtin_amdgcn_wmma_f32_16x16x32_bf16(false, a, false, b2, (short)0, acc2, false, false);
      acc3 = __builtin_amdgcn_wmma_f32_16x16x32_bf16(false, a, false, b3, (short)0, acc3, false, false);
    }
    float* orow = out + (size_t)b * NTOK * NFG_;
    const int nc0 = (nt + 0) * 16 + lrow;
    const int nc1 = (nt + 1) * 16 + lrow;
    const int nc2 = (nt + 2) * 16 + lrow;
    const int nc3 = (nt + 3) * 16 + lrow;
#pragma unroll
    for (int i = 0; i < 8; ++i) {
      int m = mbase + i;
      if (m < NTOK) {
        orow[(size_t)m * NFG_ + nc0] = fmaxf(acc0[i], 0.f);
        orow[(size_t)m * NFG_ + nc1] = fmaxf(acc1[i], 0.f);
        orow[(size_t)m * NFG_ + nc2] = fmaxf(acc2[i], 0.f);
        orow[(size_t)m * NFG_ + nc3] = fmaxf(acc3[i], 0.f);
      }
    }
  }
}

extern "C" void kernel_launch(void* const* d_in, const int* in_sizes, int n_in,
                              void* d_out, int out_size, void* d_ws, size_t ws_size,
                              hipStream_t stream) {
  (void)in_sizes; (void)n_in; (void)out_size; (void)ws_size;
  const float* x     = (const float*)d_in[0];  // [4096,12,1024]
  const float* boxes = (const float*)d_in[1];  // [49152,4]
  const float* Wt    = (const float*)d_in[2];  // [256,1024]
  const float* bth   = (const float*)d_in[3];  // [256]
  const float* Wp    = (const float*)d_in[4];  // [256,1024]
  const float* bph   = (const float*)d_in[5];  // [256]
  const float* Wg    = (const float*)d_in[6];  // [1024,1024]

  __bf16* wcat = (__bf16*)d_ws;                                   // [512,1024] bf16 (1 MB)
  __bf16* wgb  = (__bf16*)((char*)d_ws + (size_t)512 * 1024 * 2); // [1024,1024] bf16 (2 MB)

  float* out   = (float*)d_out;                    // [4096,12,1024]
  float* rgout = out + (size_t)BT * NTOK * NFG_;   // [4096,12,12]

  wconv_kernel<<<4096, 256, 0, stream>>>(Wt, Wp, Wg, wcat, wgb);
  gcn_fused_kernel<<<BT, 512, 0, stream>>>(x, boxes, bth, bph, wcat, wgb, out, rgout);
}
